// OnnxAttCov_17489106829846
// MI455X (gfx1250) — matrix operand order repaired and checked by reading, not compiled, not run on hardware
//
#include <hip/hip_runtime.h>
#include <hip/hip_bf16.h>

// ---------- CDNA5 vector types ----------
typedef __attribute__((ext_vector_type(16))) _Float16 v16h;
typedef __attribute__((ext_vector_type(8)))  _Float16 v8h;
typedef __attribute__((ext_vector_type(4)))  _Float16 v4h;
typedef __attribute__((ext_vector_type(8)))  float    v8f;

#define L_LEN   8192
#define ADIM    1024
#define DUNITS  1024
#define N_PREV  8
#define SCALING 2.0f

// -------------------------------------------------------------------------
// Kernel 1: base[a] = dec_z . mlp_dec_w[a,:] + mlp_dec_b[a] + wvec_b[a]
// wave-per-row, coalesced along d, wave32 shuffle reduce.
// -------------------------------------------------------------------------
__global__ void base_kernel(const float* __restrict__ dec_z,
                            const float* __restrict__ mlp_dec_w,
                            const float* __restrict__ mlp_dec_b,
                            const float* __restrict__ wvec_b,
                            float* __restrict__ base) {
    int wave = threadIdx.x >> 5;
    int lane = threadIdx.x & 31;
    int a = blockIdx.x * 8 + wave;           // 128 blocks * 8 waves = 1024 rows
    const float* row = mlp_dec_w + (size_t)a * DUNITS;
    float acc = 0.f;
    for (int d = lane; d < DUNITS; d += 32)
        acc = fmaf(dec_z[d], row[d], acc);
    #pragma unroll
    for (int off = 16; off > 0; off >>= 1)
        acc += __shfl_xor(acc, off, 32);
    if (lane == 0)
        base[a] = acc + mlp_dec_b[a] + wvec_b[a];
}

// -------------------------------------------------------------------------
// Kernel 2: cov[l] = sum_p att_prev[p, l]; also convert gvec_w -> f16
// -------------------------------------------------------------------------
__global__ void cov_kernel(const float* __restrict__ att_prev,
                           const float* __restrict__ gvec_w,
                           float* __restrict__ cov,
                           _Float16* __restrict__ gh) {
    int l = blockIdx.x * 256 + threadIdx.x;  // 32 blocks
    float s = 0.f;
    #pragma unroll
    for (int p = 0; p < N_PREV; ++p)
        s += att_prev[p * L_LEN + l];
    cov[l] = s;
    if (l < ADIM)
        gh[l] = (_Float16)gvec_w[l];
}

// -------------------------------------------------------------------------
// Kernel 3: e[l] = sum_a tanh(cov[l]*wv[a] + base[a] + pre[l,a]) * g[a]
//                  + gvec_b + mask[l]
// Block = 256 threads (8 waves), handles 16 rows x full K=1024.
// Staging: coalesced float4 global loads -> tanh -> f16 into padded LDS.
// Compute: each wave owns a 128-wide K slice, 4x v_wmma_f32_16x16x32_f16
//          with B = gvec broadcast over N, then cross-wave LDS reduce.
// -------------------------------------------------------------------------
#define ROWS 16
#define PADK 1032      // 1024 + 8 halves: breaks LDS bank conflicts, keeps 16B align

__global__ __launch_bounds__(256) void e_kernel(
        const float* __restrict__ pre,     // [8192,1024]
        const float* __restrict__ cov,     // [8192]
        const float* __restrict__ wv,      // wvec_w flat [1024]
        const float* __restrict__ base,    // [1024]
        const _Float16* __restrict__ gh,   // [1024] f16
        const float* __restrict__ maskv,   // [8192]
        const float* __restrict__ gvec_b,  // [1]
        float* __restrict__ e_out) {       // [8192]
    __shared__ __align__(32) _Float16 tt[ROWS * PADK];  // tanh tile, f16
    __shared__ __align__(32) _Float16 gsh[ADIM];
    __shared__ float covs[ROWS];
    __shared__ float partialv[8][ROWS];

    const int tid = threadIdx.x;
    const int rowBase = blockIdx.x * ROWS;             // 512 blocks

    if (tid < ROWS) covs[tid] = cov[rowBase + tid];
    // stage gvec (f16) into LDS
    for (int i = tid; i < ADIM / 8; i += 256)
        *(v8h*)(gsh + i * 8) = *(const v8h*)(gh + i * 8);
    __syncthreads();

    // ---- staging: 16 rows x 1024 cols, float4 per thread-iter, coalesced ----
    #pragma unroll
    for (int it = 0; it < (ROWS * ADIM / 4) / 256; ++it) {   // 16 iters
        int flat4 = it * 256 + tid;
        int r  = flat4 >> 8;                  // / (1024/4)
        int a4 = (flat4 & 255) * 4;
        float4 p4 = *(const float4*)(pre + (size_t)(rowBase + r) * ADIM + a4);
        float4 w4 = *(const float4*)(wv + a4);
        float4 b4 = *(const float4*)(base + a4);
        float cr = covs[r];
        v4h h;
        h[0] = (_Float16)tanhf(fmaf(cr, w4.x, b4.x) + p4.x);
        h[1] = (_Float16)tanhf(fmaf(cr, w4.y, b4.y) + p4.y);
        h[2] = (_Float16)tanhf(fmaf(cr, w4.z, b4.z) + p4.z);
        h[3] = (_Float16)tanhf(fmaf(cr, w4.w, b4.w) + p4.w);
        *(v4h*)(tt + r * PADK + a4) = h;
    }
    __syncthreads();

    // ---- WMMA phase: wave w handles K in [w*128, w*128+128) ----
    const int wave = tid >> 5;
    const int lane = tid & 31;
    const int row  = lane & 15;       // A-matrix M row for this lane
    const int hi   = lane >> 4;       // 0: K 0-7/16-23 halves; 1: K 8-15/24-31
    const int kw   = wave * 128;

    v8f acc = {};
    #pragma unroll
    for (int j = 0; j < 4; ++j) {
        int k0 = kw + 32 * j;
        // A: t[row, k0 + 8*hi .. +7] and t[row, k0+16+8*hi .. +7]
        const _Float16* ap = tt + row * PADK + k0 + 8 * hi;
        v8h a0 = *(const v8h*)(ap);
        v8h a1 = *(const v8h*)(ap + 16);
        v16h A;
        #pragma unroll
        for (int q = 0; q < 8; ++q) { A[q] = a0[q]; A[q + 8] = a1[q]; }
        // B[k][n] = g[k] broadcast over n: lane holds K = k0+8*hi .. +7 and k0+16+8*hi .. +7
        v8h b0 = *(const v8h*)(gsh + k0 + 8 * hi);
        v8h b1 = *(const v8h*)(gsh + k0 + 8 * hi + 16);
        v16h B;
        #pragma unroll
        for (int q = 0; q < 8; ++q) { B[q] = b0[q]; B[q + 8] = b1[q]; }
        acc = __builtin_amdgcn_wmma_f32_16x16x32_f16(
            false, A, false, B, (short)0, acc, false, false);
    }
    // D[m][n] identical over n; lane 0 holds e_0..7, lane 16 holds e_8..15
    if ((lane & 15) == 0) {
        #pragma unroll
        for (int r2 = 0; r2 < 8; ++r2)
            partialv[wave][hi * 8 + r2] = acc[r2];
    }
    __syncthreads();
    if (tid < ROWS) {
        float s = 0.f;
        #pragma unroll
        for (int w2 = 0; w2 < 8; ++w2) s += partialv[w2][tid];
        int l = rowBase + tid;
        e_out[l] = s + gvec_b[0] + maskv[l];
    }
}

// -------------------------------------------------------------------------
// Kernel 4: w = softmax(SCALING * e) over 8192, single block of 1024.
// -------------------------------------------------------------------------
__global__ __launch_bounds__(1024) void softmax_kernel(
        const float* __restrict__ e, float* __restrict__ wout) {
    __shared__ float red[1024];
    int tid = threadIdx.x;
    float v[8];
    float m = -3.0e30f;
    #pragma unroll
    for (int i = 0; i < 8; ++i) { v[i] = e[i * 1024 + tid]; m = fmaxf(m, v[i]); }
    red[tid] = m; __syncthreads();
    for (int s = 512; s > 0; s >>= 1) {
        if (tid < s) red[tid] = fmaxf(red[tid], red[tid + s]);
        __syncthreads();
    }
    m = red[0]; __syncthreads();
    float sum = 0.f;
    #pragma unroll
    for (int i = 0; i < 8; ++i) { v[i] = expf(SCALING * (v[i] - m)); sum += v[i]; }
    red[tid] = sum; __syncthreads();
    for (int s = 512; s > 0; s >>= 1) {
        if (tid < s) red[tid] += red[tid + s];
        __syncthreads();
    }
    float inv = 1.0f / red[0];
    #pragma unroll
    for (int i = 0; i < 8; ++i) wout[i * 1024 + tid] = v[i] * inv;
}

// -------------------------------------------------------------------------
// Kernel 5: copy att_prev (8x8192) into output att region
// -------------------------------------------------------------------------
__global__ void copy_kernel(const float* __restrict__ src, float* __restrict__ dst) {
    int idx = blockIdx.x * 256 + threadIdx.x;   // 256 blocks -> 65536
    dst[idx] = src[idx];
}

// -------------------------------------------------------------------------
// Kernels 6/7: c[a] = sum_l w[l] * enc_h[l,a], deterministic two-phase.
// -------------------------------------------------------------------------
__global__ void ctx_partial(const float* __restrict__ enc_h,
                            const float* __restrict__ w,
                            float* __restrict__ partials) {
    int a  = blockIdx.x * 256 + threadIdx.x;    // grid.x = 4
    int l0 = blockIdx.y * 256;                  // grid.y = 32
    float acc = 0.f;
    for (int l = l0; l < l0 + 256; ++l)
        acc = fmaf(w[l], enc_h[(size_t)l * ADIM + a], acc);
    partials[blockIdx.y * ADIM + a] = acc;
}

__global__ void ctx_reduce(const float* __restrict__ partials,
                           float* __restrict__ c) {
    int a = blockIdx.x * 256 + threadIdx.x;     // grid.x = 4
    float s = 0.f;
    #pragma unroll
    for (int y = 0; y < 32; ++y) s += partials[y * ADIM + a];
    c[a] = s;
}

// -------------------------------------------------------------------------
extern "C" void kernel_launch(void* const* d_in, const int* in_sizes, int n_in,
                              void* d_out, int out_size, void* d_ws, size_t ws_size,
                              hipStream_t stream) {
    const float* dec_z     = (const float*)d_in[0];
    const float* att_prev  = (const float*)d_in[1];
    const float* pre       = (const float*)d_in[2];
    const float* enc_h     = (const float*)d_in[3];
    const float* maskv     = (const float*)d_in[4];
    const float* wvec_w    = (const float*)d_in[5];
    const float* wvec_b    = (const float*)d_in[6];
    const float* mlp_dec_w = (const float*)d_in[7];
    const float* mlp_dec_b = (const float*)d_in[8];
    const float* gvec_w    = (const float*)d_in[9];
    const float* gvec_b    = (const float*)d_in[10];

    float* out = (float*)d_out;
    // output layout: c [1024] | att_prev copy [8*8192] | w [8192]
    float* out_c   = out;
    float* out_att = out + ADIM;
    float* out_w   = out + ADIM + N_PREV * L_LEN;

    // workspace layout (floats)
    float* basev    = (float*)d_ws;            // 1024
    float* covv     = basev + ADIM;            // 8192
    float* ev       = covv + L_LEN;            // 8192
    float* partials = ev + L_LEN;              // 32*1024
    _Float16* gh    = (_Float16*)(partials + 32 * ADIM);  // 1024 halves

    base_kernel<<<128, 256, 0, stream>>>(dec_z, mlp_dec_w, mlp_dec_b, wvec_b, basev);
    cov_kernel<<<32, 256, 0, stream>>>(att_prev, gvec_w, covv, gh);
    e_kernel<<<L_LEN / ROWS, 256, 0, stream>>>(pre, covv, wvec_w, basev, gh,
                                               maskv, gvec_b, ev);
    softmax_kernel<<<1, 1024, 0, stream>>>(ev, out_w);
    copy_kernel<<<256, 256, 0, stream>>>(att_prev, out_att);
    ctx_partial<<<dim3(4, 32), 256, 0, stream>>>(enc_h, out_w, partials);
    ctx_reduce<<<4, 256, 0, stream>>>(partials, out_c);
}
